// CuboidSelfAttention_9869834846305
// MI455X (gfx1250) — matrix operand order repaired and checked by reading, not compiled, not run on hardware
//
#include <hip/hip_runtime.h>
#include <hip/hip_bf16.h>

// ---------------------------------------------------------------------------
// CuboidSelfAttention for MI455X (gfx1250, wave32, WMMA 16x16x32 f16->f32)
// B=2 T=8 H=W=112 C=256 HEADS=8 dh=32; cuboid 2x7x7 -> cv=98 tokens
// ---------------------------------------------------------------------------

#define Bb 2
#define Tt 8
#define Hh 112
#define Ww 112
#define Cc 256
#define HEADS 8
#define DH 32
#define CTc 2
#define CHc 7
#define CWc 7
#define NTc 4
#define NHc 16
#define NWc 16
#define NOC 1024            // 4*16*16 cuboids per batch
#define CV 98               // tokens per cuboid
#define MP 112              // padded query rows (7 tiles of 16)
#define KP 128              // padded key rows (8 tiles of 16, 4 chunks of 32)
#define BN (Bb * NOC)       // 2048 cuboids total

typedef __attribute__((ext_vector_type(16))) _Float16 v16h;
typedef __attribute__((ext_vector_type(8)))  _Float16 h8;
typedef __attribute__((ext_vector_type(8)))  float    v8f;
typedef int v4i __attribute__((ext_vector_type(4)));
typedef __attribute__((address_space(1))) v4i gv4i;   // global b128 payload
typedef __attribute__((address_space(3))) v4i lv4i;   // LDS b128 payload

// ---------------------------------------------------------------------------
// CDNA5 async global->LDS copy (GLOBAL_LOAD_ASYNC_TO_LDS_B128, ASYNCcnt),
// with a portable fallback through VGPRs when the builtin is absent.
// ---------------------------------------------------------------------------
#if __has_builtin(__builtin_amdgcn_global_load_async_to_lds_b128)
#define USE_ASYNC_LDS 1
#else
#define USE_ASYNC_LDS 0
#endif

__device__ __forceinline__ void cp_b128(const _Float16* g, _Float16* l) {
#if USE_ASYNC_LDS
    __builtin_amdgcn_global_load_async_to_lds_b128((gv4i*)g, (lv4i*)l, 0, 0);
#else
    *(h8*)l = *(const h8*)g;
#endif
}

__device__ __forceinline__ void wait_async0() {
#if USE_ASYNC_LDS
#if __has_builtin(__builtin_amdgcn_s_wait_asynccnt)
    __builtin_amdgcn_s_wait_asynccnt(0);
#else
    asm volatile("s_wait_asynccnt 0x0" ::: "memory");
#endif
#endif
}

__device__ __forceinline__ v8f zero8f() {
    v8f z;
#pragma unroll
    for (int i = 0; i < 8; ++i) z[i] = 0.0f;
    return z;
}

__device__ __forceinline__ h8 zero8h() {
    h8 z;
#pragma unroll
    for (int i = 0; i < 8; ++i) z[i] = (_Float16)0.0f;
    return z;
}

// Build a 16-half fragment from two 16-byte chunks.
__device__ __forceinline__ v16h load_frag2(const _Float16* p0, const _Float16* p1) {
    h8 lo = *(const h8*)p0;
    h8 hi = *(const h8*)p1;
    return __builtin_shufflevector(lo, hi, 0,1,2,3,4,5,6,7,8,9,10,11,12,13,14,15);
}

// Table-order 16-half fragment (chunks at p and p+16; for B operands loaded from B^T row-major).
__device__ __forceinline__ v16h load_frag16(const _Float16* p) {
    return load_frag2(p, p + 16);
}

__device__ __forceinline__ v8f wmma_f16(v16h a, v16h b, v8f c) {
    return __builtin_amdgcn_wmma_f32_16x16x32_f16(false, a, false, b, (short)0, c,
                                                  false, false);
}

// ---------------------------------------------------------------------------
// K0: convert weights to f16, transposed (B^T row-major: N x K)
// ---------------------------------------------------------------------------
__global__ void k0_convert_weights(const float* __restrict__ wqkv,
                                   const float* __restrict__ wproj,
                                   _Float16* __restrict__ wqkvT,
                                   _Float16* __restrict__ wprojT) {
    int i = blockIdx.x * blockDim.x + threadIdx.x;
    if (i < 768 * 256) {
        int n = i / 256, k = i % 256;
        wqkvT[i] = (_Float16)wqkv[k * 768 + n];
    }
    if (i < 256 * 256) {
        int n = i / 256, k = i % 256;
        wprojT[i] = (_Float16)wproj[k * 256 + n];
    }
}

// ---------------------------------------------------------------------------
// K1: LayerNorm over C + cuboid reorder, f32 -> f16.  One wave per output row.
// Output: xr[bn][row(0..111)][256], rows >= 98 zeroed.
// ---------------------------------------------------------------------------
__global__ void __launch_bounds__(256) k1_ln_reorder(const float* __restrict__ x,
                                                     const float* __restrict__ gamma,
                                                     const float* __restrict__ beta,
                                                     _Float16* __restrict__ xr) {
    int gtid = blockIdx.x * blockDim.x + threadIdx.x;
    int wave = gtid >> 5;
    int lane = gtid & 31;
    int row  = wave % MP;
    int bn   = wave / MP;
    if (bn >= BN) return;

    _Float16* dst = xr + ((size_t)bn * MP + row) * Cc;
    if (row >= CV) {
        ((h8*)dst)[lane] = zero8h();
        return;
    }
    int b  = bn / NOC, n = bn % NOC;
    int ti = n / (NHc * NWc), hi = (n / NWc) % NHc, wi = n % NWc;
    int tc = row / (CHc * CWc), hc = (row / CWc) % CHc, wc = row % CWc;
    int t = ti * CTc + tc, h = hi * CHc + hc, w = wi * CWc + wc;
    const float* src = x + ((((size_t)b * Tt + t) * Hh + h) * Ww + w) * Cc;

    float4 v0 = ((const float4*)src)[lane * 2];
    float4 v1 = ((const float4*)src)[lane * 2 + 1];
    float s  = v0.x + v0.y + v0.z + v0.w + v1.x + v1.y + v1.z + v1.w;
    float sq = v0.x*v0.x + v0.y*v0.y + v0.z*v0.z + v0.w*v0.w +
               v1.x*v1.x + v1.y*v1.y + v1.z*v1.z + v1.w*v1.w;
#pragma unroll
    for (int m = 16; m >= 1; m >>= 1) {
        s  += __shfl_xor(s, m);
        sq += __shfl_xor(sq, m);
    }
    float mean = s * (1.0f / Cc);
    float var  = sq * (1.0f / Cc) - mean * mean;
    float rstd = rsqrtf(var + 1e-6f);

    float4 g0 = ((const float4*)gamma)[lane * 2];
    float4 g1 = ((const float4*)gamma)[lane * 2 + 1];
    float4 bb0 = ((const float4*)beta)[lane * 2];
    float4 bb1 = ((const float4*)beta)[lane * 2 + 1];

    h8 o;
    o[0] = (_Float16)((v0.x - mean) * rstd * g0.x + bb0.x);
    o[1] = (_Float16)((v0.y - mean) * rstd * g0.y + bb0.y);
    o[2] = (_Float16)((v0.z - mean) * rstd * g0.z + bb0.z);
    o[3] = (_Float16)((v0.w - mean) * rstd * g0.w + bb0.w);
    o[4] = (_Float16)((v1.x - mean) * rstd * g1.x + bb1.x);
    o[5] = (_Float16)((v1.y - mean) * rstd * g1.y + bb1.y);
    o[6] = (_Float16)((v1.z - mean) * rstd * g1.z + bb1.z);
    o[7] = (_Float16)((v1.w - mean) * rstd * g1.w + bb1.w);
    ((h8*)dst)[lane] = o;
}

// ---------------------------------------------------------------------------
// K2: QKV GEMM per cuboid.  M=112, N=768, K=256.  8 waves x 6 N-tiles each.
// Epilogue scatters: Q row-major (112x32, scaled dh^-1/2), K row-major
// (128x32, rows 112..127 zeroed), V transposed (32x128, cols 112..127 zeroed).
// ---------------------------------------------------------------------------
__global__ void __launch_bounds__(256) k2_qkv(const _Float16* __restrict__ xr,
                                              const _Float16* __restrict__ wqkvT,
                                              _Float16* __restrict__ qb,
                                              _Float16* __restrict__ kb,
                                              _Float16* __restrict__ vtb) {
    __shared__ __align__(16) _Float16 xs[MP * Cc];     // 112x256 f16 = 56 KB
    int bn = blockIdx.x;
    int tid = threadIdx.x, lane = tid & 31, wv = tid >> 5;
    int l16 = lane & 15, hs = lane >> 4;

    const _Float16* src = xr + (size_t)bn * MP * Cc;
#pragma unroll 4
    for (int i = tid; i < MP * Cc / 8; i += 256) cp_b128(src + i * 8, xs + i * 8);
    wait_async0();
    __syncthreads();

#pragma unroll 1
    for (int it = 0; it < 6; ++it) {
        int ntile = wv * 6 + it;
        int cg = ntile * 16;
        __builtin_prefetch(wqkvT + (size_t)(cg + l16) * Cc, 0, 1);
        v8f acc[7];
#pragma unroll
        for (int m = 0; m < 7; ++m) acc[m] = zero8f();

#pragma unroll 1
        for (int kc = 0; kc < 8; ++kc) {
            const _Float16* bp = wqkvT + (size_t)(cg + l16) * Cc + kc * 32 + hs * 8;
            v16h bfrag = load_frag16(bp);
#pragma unroll
            for (int mt = 0; mt < 7; ++mt) {
                const _Float16* ap = xs + (mt * 16 + l16) * Cc + kc * 32 + hs * 8;
                v16h afrag = load_frag2(ap, ap + 16);
                acc[mt] = wmma_f16(afrag, bfrag, acc[mt]);
            }
        }

        int kind = cg >> 8;                // 0=q 1=k 2=v
        int head = (cg >> 5) & 7;
        int dbase = cg & 31;
        size_t bh = (size_t)bn * HEADS + head;
        if (kind == 0) {
            const float qs = 0.17677669529663687f;   // 32^-0.5
#pragma unroll
            for (int mt = 0; mt < 7; ++mt)
#pragma unroll
                for (int j = 0; j < 8; ++j) {
                    int row = mt * 16 + j + 8 * hs;
                    qb[(bh * MP + row) * DH + dbase + l16] =
                        (_Float16)(acc[mt][j] * qs);
                }
        } else if (kind == 1) {
#pragma unroll
            for (int mt = 0; mt < 7; ++mt)
#pragma unroll
                for (int j = 0; j < 8; ++j) {
                    int row = mt * 16 + j + 8 * hs;
                    kb[(bh * KP + row) * DH + dbase + l16] = (_Float16)acc[mt][j];
                }
        } else {
            int d = dbase + l16;
#pragma unroll
            for (int mt = 0; mt < 7; ++mt) {
                h8 o;
#pragma unroll
                for (int j = 0; j < 8; ++j) o[j] = (_Float16)acc[mt][j];
                _Float16* vp = vtb + (bh * DH + d) * KP + mt * 16 + 8 * hs;
                *(h8*)vp = o;
            }
        }
    }

    // zero K pad rows 112..127 (8 heads x 16 rows x 32 halfs)
    h8 z = zero8h();
    for (int i = tid; i < HEADS * 16 * DH / 8; i += 256) {
        int head = i / (16 * DH / 8);
        int off  = i % (16 * DH / 8);
        h8* p = (h8*)(kb + (((size_t)bn * HEADS + head) * KP + MP) * DH);
        p[off] = z;
    }
    // zero V^T pad cols 112..127 (8 heads x 32 rows x 16 halfs)
    for (int i = tid; i < HEADS * DH * 2; i += 256) {
        int head = i / (DH * 2);
        int rr = i % (DH * 2);
        int row = rr >> 1, half = rr & 1;
        h8* p = (h8*)(vtb + (((size_t)bn * HEADS + head) * DH + row) * KP + MP);
        p[half] = z;
    }
}

// ---------------------------------------------------------------------------
// K3: attention.  One wave per (cuboid, head).  K/V^T staged in LDS via
// async global->LDS copies.  Per 16-query tile: 8 score WMMAs -> masked
// softmax -> attn f16 in LDS -> 2x4 WMMAs against V^T.
// Output: obuf[bn][112][256] f16.
// ---------------------------------------------------------------------------
__global__ void __launch_bounds__(256) k3_attn(const _Float16* __restrict__ qb,
                                               const _Float16* __restrict__ kb,
                                               const _Float16* __restrict__ vtb,
                                               _Float16* __restrict__ obuf) {
    __shared__ __align__(16) _Float16 ks[HEADS][KP * DH];   // 64 KB
    __shared__ __align__(16) _Float16 vs[HEADS][DH * KP];   // 64 KB
    __shared__ __align__(16) _Float16 as[HEADS][16 * KP];   // 32 KB
    int bn = blockIdx.x;
    int tid = threadIdx.x, lane = tid & 31, head = tid >> 5;
    int l16 = lane & 15, hs = lane >> 4;
    size_t bh = (size_t)bn * HEADS + head;

    {   // per-wave stage of K and V^T into this wave's LDS region (ASYNCcnt)
        const _Float16* kg = kb + bh * KP * DH;
        const _Float16* vg = vtb + bh * DH * KP;
#pragma unroll 4
        for (int i = lane; i < KP * DH / 8; i += 32)
            cp_b128(kg + i * 8, ks[head] + i * 8);
#pragma unroll 4
        for (int i = lane; i < DH * KP / 8; i += 32)
            cp_b128(vg + i * 8, vs[head] + i * 8);
        wait_async0();
    }

    const float NEG = -1e30f;
#pragma unroll 1
    for (int mt = 0; mt < 7; ++mt) {
        const _Float16* qp = qb + (bh * MP + mt * 16 + l16) * DH + hs * 8;
        v16h aq = load_frag2(qp, qp + 16);

        v8f s[8];
#pragma unroll
        for (int nt = 0; nt < 8; ++nt) s[nt] = zero8f();
#pragma unroll
        for (int nt = 0; nt < 8; ++nt) {
            const _Float16* kp = ks[head] + (nt * 16 + l16) * DH + hs * 8;
            v16h bk = load_frag16(kp);
            s[nt] = wmma_f16(aq, bk, s[nt]);
        }

        // mask padded keys (key index = nt*16 + l16)
#pragma unroll
        for (int nt = 0; nt < 8; ++nt)
            if (nt * 16 + l16 >= CV) {
#pragma unroll
                for (int j = 0; j < 8; ++j) s[nt][j] = NEG;
            }

        // softmax per query row; rows spread over nt tiles, keys over 16 lanes
#pragma unroll
        for (int j = 0; j < 8; ++j) {
            float mx = NEG;
#pragma unroll
            for (int nt = 0; nt < 8; ++nt) mx = fmaxf(mx, s[nt][j]);
#pragma unroll
            for (int m = 8; m >= 1; m >>= 1) mx = fmaxf(mx, __shfl_xor(mx, m));
            float sum = 0.0f;
#pragma unroll
            for (int nt = 0; nt < 8; ++nt) {
                float p = __expf(s[nt][j] - mx);
                s[nt][j] = p;
                sum += p;
            }
#pragma unroll
            for (int m = 8; m >= 1; m >>= 1) sum += __shfl_xor(sum, m);
            float r = 1.0f / sum;
#pragma unroll
            for (int nt = 0; nt < 8; ++nt) s[nt][j] *= r;
        }

        // stage attn tile (16 x 128) as f16 row-major in LDS
#pragma unroll
        for (int nt = 0; nt < 8; ++nt)
#pragma unroll
            for (int j = 0; j < 8; ++j)
                as[head][(j + 8 * hs) * KP + nt * 16 + l16] = (_Float16)s[nt][j];

        // O tile = attn(16x128) @ V(128x32)
#pragma unroll
        for (int nt2 = 0; nt2 < 2; ++nt2) {
            v8f oacc = zero8f();
#pragma unroll
            for (int kc = 0; kc < 4; ++kc) {
                const _Float16* ap = as[head] + l16 * KP + kc * 32 + hs * 8;
                v16h aa = load_frag2(ap, ap + 16);
                const _Float16* vp = vs[head] + (nt2 * 16 + l16) * KP + kc * 32 + hs * 8;
                v16h bv = load_frag16(vp);
                oacc = wmma_f16(aa, bv, oacc);
            }
            int c = head * DH + nt2 * 16 + l16;
#pragma unroll
            for (int j = 0; j < 8; ++j) {
                int q = mt * 16 + j + 8 * hs;
                obuf[((size_t)bn * MP + q) * Cc + c] = (_Float16)oacc[j];
            }
        }
    }
}

// ---------------------------------------------------------------------------
// K4: output projection (112x256 @ 256x256 + bias) + reverse cuboid scatter.
// ---------------------------------------------------------------------------
__global__ void __launch_bounds__(256) k4_proj(const _Float16* __restrict__ obuf,
                                               const _Float16* __restrict__ wprojT,
                                               const float* __restrict__ bias,
                                               float* __restrict__ out) {
    __shared__ __align__(16) _Float16 os[MP * Cc];     // 56 KB
    int bn = blockIdx.x;
    int tid = threadIdx.x, lane = tid & 31, wv = tid >> 5;
    int l16 = lane & 15, hs = lane >> 4;

    const _Float16* src = obuf + (size_t)bn * MP * Cc;
#pragma unroll 4
    for (int i = tid; i < MP * Cc / 8; i += 256) cp_b128(src + i * 8, os + i * 8);
    wait_async0();
    __syncthreads();

    int b  = bn / NOC, n = bn % NOC;
    int ti = n / (NHc * NWc), hi = (n / NWc) % NHc, wi = n % NWc;

#pragma unroll 1
    for (int it = 0; it < 2; ++it) {
        int ntile = wv * 2 + it;
        int cg = ntile * 16;
        __builtin_prefetch(wprojT + (size_t)(cg + l16) * Cc, 0, 1);
        v8f acc[7];
#pragma unroll
        for (int m = 0; m < 7; ++m) acc[m] = zero8f();

#pragma unroll 1
        for (int kc = 0; kc < 8; ++kc) {
            const _Float16* bp = wprojT + (size_t)(cg + l16) * Cc + kc * 32 + hs * 8;
            v16h bfrag = load_frag16(bp);
#pragma unroll
            for (int mt = 0; mt < 7; ++mt) {
                const _Float16* ap = os + (mt * 16 + l16) * Cc + kc * 32 + hs * 8;
                v16h afrag = load_frag2(ap, ap + 16);
                acc[mt] = wmma_f16(afrag, bfrag, acc[mt]);
            }
        }

        int c = cg + l16;
        float bc = bias[c];
#pragma unroll
        for (int mt = 0; mt < 7; ++mt)
#pragma unroll
            for (int j = 0; j < 8; ++j) {
                int q = mt * 16 + j + 8 * hs;
                if (q < CV) {
                    int tc = q / (CHc * CWc), hc = (q / CWc) % CHc, wc = q % CWc;
                    int t = ti * CTc + tc, h = hi * CHc + hc, w = wi * CWc + wc;
                    out[((((size_t)b * Tt + t) * Hh + h) * Ww + w) * Cc + c] =
                        acc[mt][j] + bc;
                }
            }
    }
}

// ---------------------------------------------------------------------------
extern "C" void kernel_launch(void* const* d_in, const int* in_sizes, int n_in,
                              void* d_out, int out_size, void* d_ws, size_t ws_size,
                              hipStream_t stream) {
    const float* x        = (const float*)d_in[0];
    const float* ln_scale = (const float*)d_in[1];
    const float* ln_bias  = (const float*)d_in[2];
    const float* w_qkv    = (const float*)d_in[3];
    const float* w_proj   = (const float*)d_in[4];
    const float* b_proj   = (const float*)d_in[5];
    float* out = (float*)d_out;

    _Float16* base = (_Float16*)d_ws;
    size_t o = 0;
    _Float16* wqkvT  = base + o; o += (size_t)768 * 256;
    _Float16* wprojT = base + o; o += (size_t)256 * 256;
    _Float16* xr     = base + o; o += (size_t)BN * MP * Cc;
    _Float16* qb     = base + o; o += (size_t)BN * HEADS * MP * DH;
    _Float16* kb     = base + o; o += (size_t)BN * HEADS * KP * DH;
    _Float16* vtb    = base + o; o += (size_t)BN * HEADS * DH * KP;
    _Float16* obuf   = base + o; o += (size_t)BN * MP * Cc;
    (void)ws_size; (void)in_sizes; (void)n_in; (void)out_size;

    k0_convert_weights<<<(768 * 256 + 255) / 256, 256, 0, stream>>>(
        w_qkv, w_proj, wqkvT, wprojT);
    k1_ln_reorder<<<(BN * MP) / 8, 256, 0, stream>>>(x, ln_scale, ln_bias, xr);
    k2_qkv<<<BN, 256, 0, stream>>>(xr, wqkvT, qb, kb, vtb);
    k3_attn<<<BN, 256, 0, stream>>>(qb, kb, vtb, obuf);
    k4_proj<<<BN, 256, 0, stream>>>(obuf, wprojT, b_proj, out);
}
